// GATMutPPI_11132555231391
// MI455X (gfx1250) — matrix operand order repaired and checked by reading, not compiled, not run on hardware
//
#include <hip/hip_runtime.h>
#include <hip/hip_bf16.h>
#include <math.h>

typedef __attribute__((ext_vector_type(16))) _Float16 v16h;
typedef __attribute__((ext_vector_type(8)))  _Float16 v8h;
typedef __attribute__((ext_vector_type(8)))  float    v8f;

#define NN      20000
#define E_RAW   200000
#define E_TOT   (E_RAW + NN)
#define FIN     1280
#define HC1     1024     // 4 heads * 256
#define H1      4
#define C1      256
#define C2      128
#define NEG_SLOPE 0.2f

// ---------------------------------------------------------------- utilities
__global__ void k_cvt_f16(const float* __restrict__ in, _Float16* __restrict__ out, int n) {
  int i = blockIdx.x * blockDim.x + threadIdx.x;
  if (i < n) out[i] = (_Float16)in[i];
}

// convert + transpose: in = K x N row-major f32, out = N x K row-major f16
__global__ void k_cvt_f16_T(const float* __restrict__ in, _Float16* __restrict__ out,
                            int K, int N) {
  int i = blockIdx.x * blockDim.x + threadIdx.x;
  if (i >= K * N) return;
  int n = i / K, k = i - n * K;              // write side contiguous
  out[i] = (_Float16)in[(size_t)k * N + n];
}

__device__ __forceinline__ float lrelu(float v) { return v > 0.f ? v : NEG_SLOPE * v; }
__device__ __forceinline__ unsigned f2ord(float f) {
  unsigned u = __float_as_uint(f);
  return (u & 0x80000000u) ? ~u : (u | 0x80000000u);
}
__device__ __forceinline__ float ord2f(unsigned u) {
  return __uint_as_float((u & 0x80000000u) ? (u ^ 0x80000000u) : ~u);
}
__device__ __forceinline__ void edge_sd(const int* __restrict__ ei, int e, int& s, int& d) {
  if (e < E_RAW) { s = ei[e]; d = ei[E_RAW + e]; } else { s = d = e - E_RAW; }
}

// ---------------------------------------------------------------- WMMA GEMM
// C(MxN,f32) = A(MxK,f16,row-major) * Bt(NxK,f16,row-major  == B^T).
// K % 32 == 0, N % 128 == 0.  Both LDS tiles are K-contiguous -> all-vector LDS traffic.
#define BM 128
#define BN 128
#define BK 32

__global__ __launch_bounds__(256)
void k_gemm_wmma(const _Float16* __restrict__ A, const _Float16* __restrict__ Bt,
                 float* __restrict__ C, int M, int N, int K) {
  __shared__ _Float16 As[BM][BK + 8];   // A tile, K contiguous per row
  __shared__ _Float16 Bs[BN][BK + 8];   // B^T tile, K contiguous per column-of-B

  const int tid   = threadIdx.x;
  const int lane  = tid & 31;
  const int wid   = tid >> 5;        // 0..7
  const int waveM = wid >> 1;        // 0..3 : 32 rows each
  const int waveN = wid & 1;         // 0..1 : 64 cols each
  const int blockM = blockIdx.y * BM;
  const int blockN = blockIdx.x * BN;
  const int lrow  = lane & 15;
  const int khalf = lane >> 4;       // 0/1 : which K-half this lane holds

  const int r  = tid >> 1;           // 0..127 (A row / B column within tile)
  const int cb = (tid & 1) * 16;     // 0/16  (K chunk)

  v8f acc[2][4] = {};

  for (int k0 = 0; k0 < K; k0 += BK) {
    // --- stage A tile: one 16-half K-chunk per thread, b128 x2 both sides
    {
      int gr = blockM + r;
      uint4 u0, u1;
      if (gr < M) {
        const uint4* gp = (const uint4*)(A + (size_t)gr * K + k0 + cb);
        u0 = gp[0]; u1 = gp[1];
      } else {
        u0 = make_uint4(0,0,0,0); u1 = make_uint4(0,0,0,0);
      }
      *(uint4*)&As[r][cb]     = u0;
      *(uint4*)&As[r][cb + 8] = u1;
    }
    // --- stage B^T tile: same pattern, no transpose needed (done once up front)
    {
      const uint4* gp = (const uint4*)(Bt + (size_t)(blockN + r) * K + k0 + cb);
      uint4 u0 = gp[0], u1 = gp[1];
      *(uint4*)&Bs[r][cb]     = u0;
      *(uint4*)&Bs[r][cb + 8] = u1;
    }
    if (k0 + BK < K) { // speculative prefetch of next tiles -> global_prefetch_b8
      __builtin_prefetch(A  + (size_t)(blockM + r) * K + k0 + BK, 0, 0);
      __builtin_prefetch(Bt + (size_t)(blockN + r) * K + k0 + BK, 0, 0);
    }
    __syncthreads();

    // --- gather fragments per ISA 7.12.2 layouts: all 16B vector LDS loads
    v16h aF[2]; v16h bF[4];
    #pragma unroll
    for (int m = 0; m < 2; ++m) {
      const int row = waveM * 32 + m * 16 + lrow;
      v8h lo = *(const v8h*)&As[row][khalf * 8];
      v8h hi = *(const v8h*)&As[row][16 + khalf * 8];
      aF[m] = __builtin_shufflevector(lo, hi, 0,1,2,3,4,5,6,7,8,9,10,11,12,13,14,15);
    }
    #pragma unroll
    for (int n = 0; n < 4; ++n) {
      const int col = waveN * 64 + n * 16 + lrow;
      v8h lo = *(const v8h*)&Bs[col][khalf * 8];
      v8h hi = *(const v8h*)&Bs[col][16 + khalf * 8];
      bF[n] = __builtin_shufflevector(lo, hi, 0,1,2,3,4,5,6,7,8,9,10,11,12,13,14,15);
    }
    #pragma unroll
    for (int m = 0; m < 2; ++m)
      #pragma unroll
      for (int n = 0; n < 4; ++n)
        acc[m][n] = __builtin_amdgcn_wmma_f32_16x16x32_f16(
            false, aF[m], false, bF[n], (short)0, acc[m][n], false, false);
    __syncthreads();
  }

  // store: C/D layout — VGPR r holds M = (lane>=16 ? 8 : 0) + r, N = lane&15
  #pragma unroll
  for (int m = 0; m < 2; ++m) {
    #pragma unroll
    for (int n = 0; n < 4; ++n) {
      const int colg = blockN + waveN * 64 + n * 16 + (lane & 15);
      #pragma unroll
      for (int rr = 0; rr < 8; ++rr) {
        const int rowg = blockM + waveM * 32 + m * 16 + (lane >> 4) * 8 + rr;
        if (rowg < M) C[(size_t)rowg * N + colg] = acc[m][n][rr];
      }
    }
  }
}

// ---------------------------------------------------------------- attention scores
// one wave per (node, head): a_src/a_dst dot products over C channels
__global__ void k_att_scores(const float* __restrict__ h, const float* __restrict__ att_s,
                             const float* __restrict__ att_d, float* __restrict__ as_,
                             float* __restrict__ ad_, int Nn, int H, int Cc) {
  int wave = blockIdx.x * (blockDim.x >> 5) + (threadIdx.x >> 5);
  int lane = threadIdx.x & 31;
  if (wave >= Nn * H) return;
  int n = wave / H, hh = wave % H;
  const float* hp  = h + (size_t)n * H * Cc + (size_t)hh * Cc;
  const float* asp = att_s + (size_t)hh * Cc;
  const float* adp = att_d + (size_t)hh * Cc;
  float s0 = 0.f, s1 = 0.f;
  for (int c = lane; c < Cc; c += 32) { float v = hp[c]; s0 += v * asp[c]; s1 += v * adp[c]; }
  #pragma unroll
  for (int o = 16; o > 0; o >>= 1) { s0 += __shfl_xor(s0, o, 32); s1 += __shfl_xor(s1, o, 32); }
  if (lane == 0) { as_[wave] = s0; ad_[wave] = s1; }
}

// ---------------------------------------------------------------- edge passes
__global__ void k_edge_max(const int* __restrict__ ei, const float* __restrict__ as_,
                           const float* __restrict__ ad_, unsigned* __restrict__ segmax, int H) {
  int e = blockIdx.x * blockDim.x + threadIdx.x;
  if (e >= E_TOT) return;
  int s, d; edge_sd(ei, e, s, d);
  for (int h = 0; h < H; ++h) {
    float v = lrelu(as_[s * H + h] + ad_[d * H + h]);
    atomicMax(&segmax[d * H + h], f2ord(v));
  }
}

__global__ void k_edge_expsum(const int* __restrict__ ei, const float* __restrict__ as_,
                              const float* __restrict__ ad_, const unsigned* __restrict__ segmax,
                              float* __restrict__ segsum, int H) {
  int e = blockIdx.x * blockDim.x + threadIdx.x;
  if (e >= E_TOT) return;
  int s, d; edge_sd(ei, e, s, d);
  for (int h = 0; h < H; ++h) {
    float v = lrelu(as_[s * H + h] + ad_[d * H + h]);
    float m = ord2f(segmax[d * H + h]);
    atomicAdd(&segsum[d * H + h], __expf(v - m));
  }
}

// one block per edge; per-head softmax coefficient computed once into LDS
__global__ __launch_bounds__(256)
void k_edge_agg(const int* __restrict__ ei, const float* __restrict__ as_,
                const float* __restrict__ ad_, const unsigned* __restrict__ segmax,
                const float* __restrict__ segsum, const float* __restrict__ hsrc,
                float* __restrict__ out, int H, int cshift) {
  __shared__ float s_alpha[8];
  int e = blockIdx.x;
  int s, d; edge_sd(ei, e, s, d);
  if ((int)threadIdx.x < H) {
    int h = threadIdx.x;
    float v = lrelu(as_[s * H + h] + ad_[d * H + h]);
    float m = ord2f(segmax[d * H + h]);
    s_alpha[h] = __expf(v - m) / (segsum[d * H + h] + 1e-16f);
  }
  __syncthreads();
  const int HC = H << cshift;
  for (int c = threadIdx.x; c < HC; c += blockDim.x) {
    float a = s_alpha[c >> cshift];
    atomicAdd(&out[(size_t)d * HC + c], a * hsrc[(size_t)s * HC + c]);
  }
}

// ---------------------------------------------------------------- bias + relu (+cvt)
__global__ void k_bias_relu(const float* __restrict__ in, const float* __restrict__ bias,
                            float* __restrict__ of32, _Float16* __restrict__ of16,
                            int n, int mask) {
  int i = blockIdx.x * blockDim.x + threadIdx.x;
  if (i >= n) return;
  float v = in[i] + bias[i & mask];
  v = v > 0.f ? v : 0.f;
  if (of32) of32[i] = v;
  if (of16) of16[i] = (_Float16)v;
}

// ---------------------------------------------------------------- small MLPs
__global__ __launch_bounds__(128)
void k_mut_mlp(const float* __restrict__ msd, const float* __restrict__ W1,
               const float* __restrict__ b1, const float* __restrict__ W2,
               const float* __restrict__ b2, float* __restrict__ pmd) {
  __shared__ float p1[128];
  int t = threadIdx.x;
  float s = b1[t];
  for (int i = 0; i < 1024; ++i) s += msd[i] * W1[i * 128 + t];
  p1[t] = s > 0.f ? s : 0.f;
  __syncthreads();
  float s2 = b2[t];
  for (int i = 0; i < 128; ++i) s2 += p1[i] * W2[i * 128 + t];
  pmd[t] = s2;
}

__global__ __launch_bounds__(256)
void k_head_mlp(const float* __restrict__ feat_all, const int* __restrict__ midx,
                const float* __restrict__ pmd,
                const float* __restrict__ Wp1, const float* __restrict__ bp1,
                const float* __restrict__ Wp2, const float* __restrict__ bp2,
                const float* __restrict__ Wp3, const float* __restrict__ bp3,
                float* __restrict__ out) {
  __shared__ float z[256];
  __shared__ float z1[128];
  __shared__ float z2[32];
  int t = threadIdx.x;
  int mi = midx[0];
  z[t] = (t < 128) ? feat_all[(size_t)mi * 128 + t] : pmd[t - 128];
  __syncthreads();
  if (t < 128) {
    float s = bp1[t];
    for (int i = 0; i < 256; ++i) s += z[i] * Wp1[i * 128 + t];
    z1[t] = s > 0.f ? s : 0.f;
  }
  __syncthreads();
  if (t < 32) {
    float s = bp2[t];
    for (int i = 0; i < 128; ++i) s += z1[i] * Wp2[i * 32 + t];
    z2[t] = s > 0.f ? s : 0.f;
  }
  __syncthreads();
  if (t == 0) {
    float s = bp3[0];
    for (int i = 0; i < 32; ++i) s += z2[i] * Wp3[i];
    out[0] = s;
  }
}

// ---------------------------------------------------------------- launcher
extern "C" void kernel_launch(void* const* d_in, const int* in_sizes, int n_in,
                              void* d_out, int out_size, void* d_ws, size_t ws_size,
                              hipStream_t stream) {
  (void)in_sizes; (void)n_in; (void)out_size; (void)ws_size;

  const float* x      = (const float*)d_in[0];
  const int*   ei     = (const int*)  d_in[1];
  const int*   midx   = (const int*)  d_in[2];
  const float* msd    = (const float*)d_in[3];
  const float* W_mut1 = (const float*)d_in[4];
  const float* b_mut1 = (const float*)d_in[5];
  const float* W_mut2 = (const float*)d_in[6];
  const float* b_mut2 = (const float*)d_in[7];
  const float* g1W    = (const float*)d_in[8];
  const float* g1as   = (const float*)d_in[9];
  const float* g1ad   = (const float*)d_in[10];
  const float* g1b    = (const float*)d_in[11];
  const float* g2W    = (const float*)d_in[12];
  const float* g2as   = (const float*)d_in[13];
  const float* g2ad   = (const float*)d_in[14];
  const float* g2b    = (const float*)d_in[15];
  const float* Wp1    = (const float*)d_in[16];
  const float* bp1    = (const float*)d_in[17];
  const float* Wp2    = (const float*)d_in[18];
  const float* bp2    = (const float*)d_in[19];
  const float* Wp3    = (const float*)d_in[20];
  const float* bp3    = (const float*)d_in[21];
  float* out = (float*)d_out;

  // workspace carving (256B aligned)
  char* p = (char*)d_ws;
  auto carve = [&](size_t bytes) -> void* {
    void* r = (void*)p; p += (bytes + 255) & ~(size_t)255; return r;
  };
  _Float16* xh   = (_Float16*)carve((size_t)NN * FIN * 2);
  _Float16* w1t  = (_Float16*)carve((size_t)FIN * HC1 * 2);   // gat1_W^T, f16 (HC1 x FIN)
  float*    h1   = (float*)   carve((size_t)NN * HC1 * 4);
  float*    as1  = (float*)   carve((size_t)NN * H1 * 4);
  float*    ad1  = (float*)   carve((size_t)NN * H1 * 4);
  unsigned* m1   = (unsigned*)carve((size_t)NN * H1 * 4);
  float*    s1   = (float*)   carve((size_t)NN * H1 * 4);
  float*    o1   = (float*)   carve((size_t)NN * HC1 * 4);
  _Float16* h1h  = (_Float16*)carve((size_t)NN * HC1 * 2);
  _Float16* w2t  = (_Float16*)carve((size_t)HC1 * C2 * 2);    // gat2_W^T, f16 (C2 x HC1)
  float*    h2   = (float*)   carve((size_t)NN * C2 * 4);
  float*    as2  = (float*)   carve((size_t)NN * 4);
  float*    ad2  = (float*)   carve((size_t)NN * 4);
  unsigned* m2   = (unsigned*)carve((size_t)NN * 4);
  float*    s2   = (float*)   carve((size_t)NN * 4);
  float*    o2   = (float*)   carve((size_t)NN * C2 * 4);
  float*    fin  = (float*)   carve((size_t)NN * C2 * 4);
  float*    pmd  = (float*)   carve(512);

  const int TB = 256;
  auto blocks = [](long long n, int tb) { return (unsigned)((n + tb - 1) / tb); };

  // ---- convert inputs to f16 (weights transposed once for K-contiguous staging)
  k_cvt_f16<<<blocks((long long)NN * FIN, TB), TB, 0, stream>>>(x, xh, NN * FIN);
  k_cvt_f16_T<<<blocks((long long)FIN * HC1, TB), TB, 0, stream>>>(g1W, w1t, FIN, HC1);
  k_cvt_f16_T<<<blocks((long long)HC1 * C2, TB), TB, 0, stream>>>(g2W, w2t, HC1, C2);

  // ---- mutation MLP (independent)
  k_mut_mlp<<<1, 128, 0, stream>>>(msd, W_mut1, b_mut1, W_mut2, b_mut2, pmd);

  // ---- GAT layer 1
  {
    dim3 grid(HC1 / BN, (NN + BM - 1) / BM);
    k_gemm_wmma<<<grid, 256, 0, stream>>>(xh, w1t, h1, NN, HC1, FIN);
  }
  k_att_scores<<<blocks((long long)NN * H1 * 32, TB), TB, 0, stream>>>(
      h1, g1as, g1ad, as1, ad1, NN, H1, C1);
  hipMemsetAsync(m1, 0, (size_t)NN * H1 * 4, stream);   // ordered-uint 0 == -inf
  hipMemsetAsync(s1, 0, (size_t)NN * H1 * 4, stream);
  hipMemsetAsync(o1, 0, (size_t)NN * HC1 * 4, stream);
  k_edge_max   <<<blocks(E_TOT, TB), TB, 0, stream>>>(ei, as1, ad1, m1, H1);
  k_edge_expsum<<<blocks(E_TOT, TB), TB, 0, stream>>>(ei, as1, ad1, m1, s1, H1);
  k_edge_agg   <<<E_TOT, 256, 0, stream>>>(ei, as1, ad1, m1, s1, h1, o1, H1, 8);
  k_bias_relu<<<blocks((long long)NN * HC1, TB), TB, 0, stream>>>(
      o1, g1b, (float*)nullptr, h1h, NN * HC1, HC1 - 1);

  // ---- GAT layer 2
  {
    dim3 grid(C2 / BN, (NN + BM - 1) / BM);
    k_gemm_wmma<<<grid, 256, 0, stream>>>(h1h, w2t, h2, NN, C2, HC1);
  }
  k_att_scores<<<blocks((long long)NN * 32, TB), TB, 0, stream>>>(
      h2, g2as, g2ad, as2, ad2, NN, 1, C2);
  hipMemsetAsync(m2, 0, (size_t)NN * 4, stream);
  hipMemsetAsync(s2, 0, (size_t)NN * 4, stream);
  hipMemsetAsync(o2, 0, (size_t)NN * C2 * 4, stream);
  k_edge_max   <<<blocks(E_TOT, TB), TB, 0, stream>>>(ei, as2, ad2, m2, 1);
  k_edge_expsum<<<blocks(E_TOT, TB), TB, 0, stream>>>(ei, as2, ad2, m2, s2, 1);
  k_edge_agg   <<<E_TOT, 256, 0, stream>>>(ei, as2, ad2, m2, s2, h2, o2, 1, 7);
  k_bias_relu<<<blocks((long long)NN * C2, TB), TB, 0, stream>>>(
      o2, g2b, fin, (_Float16*)nullptr, NN * C2, C2 - 1);

  // ---- head MLP
  k_head_mlp<<<1, 256, 0, stream>>>(fin, midx, pmd, Wp1, bp1, Wp2, bp2, Wp3, bp3, out);
}
